// DynamicRNN_89996744721167
// MI455X (gfx1250) — matrix operand order repaired and checked by reading, not compiled, not run on hardware
//
#include <hip/hip_runtime.h>

// Problem constants (match reference)
#define B_   128
#define T_   2048
#define F_   3
#define U_   256
#define G_   1024          // 4*U
#define NWG  16            // persistent workgroups; each owns 16 units
#define WAVES 4
#define TPB  (WAVES * 32)  // 128 threads (wave32)
#define KT   8             // 256 / 32 K-tiles

typedef __attribute__((ext_vector_type(16))) __bf16 v16bf;
typedef __attribute__((ext_vector_type(8)))  __bf16 v8bf;
typedef __attribute__((ext_vector_type(8)))  float  v8f;

// ---- fast transcendentals (v_exp_f32 / v_rcp_f32) --------------------------
__device__ __forceinline__ float fast_exp(float x) {
    return __builtin_amdgcn_exp2f(x * 1.44269504088896f);
}
__device__ __forceinline__ float fast_sigmoid(float x) {
    return __builtin_amdgcn_rcpf(1.0f + fast_exp(-x));
}
__device__ __forceinline__ float fast_tanh(float x) {
    x = fmaxf(fminf(x, 10.0f), -10.0f);          // avoid inf/inf
    float e = fast_exp(-2.0f * x);
    return (1.0f - e) * __builtin_amdgcn_rcpf(1.0f + e);
}

// ---- one-time Wh repack: f32 row-major [256,1024] -> bf16 WMMA-B blocks ----
// Block (w, gate, kt) is 512 halves: lane l, elem e -> Wh[kt*32 + (l>>4)*16 + e,
//                                                        gate*256 + w*16 + (l&15)]
__global__ void pack_Wh(const float* __restrict__ Wh, __bf16* __restrict__ whp) {
    int p = blockIdx.x * blockDim.x + threadIdx.x;        // 0 .. 262143
    if (p >= NWG * 4 * KT * 512) return;
    int e    = p & 15;
    int lane = (p >> 4) & 31;
    int blk  = p >> 9;                                    // 0 .. 511
    int kt   = blk & 7;
    int gt   = (blk >> 3) & 3;
    int w    = blk >> 5;
    int K = kt * 32 + ((lane >> 4) << 4) + e;
    int N = gt * U_ + w * 16 + (lane & 15);
    whp[p] = (__bf16)Wh[K * G_ + N];
}

// ---- per-launch workspace init: h0 = 0, barrier = 0 ------------------------
__global__ void init_ws(__bf16* __restrict__ hbuf0, unsigned* __restrict__ bar) {
    int i = blockIdx.x * blockDim.x + threadIdx.x;
    if (i < B_ * U_) hbuf0[i] = (__bf16)0.0f;
    if (i == 0) *bar = 0u;
}

// ---- persistent fused LSTM -------------------------------------------------
__global__ __launch_bounds__(TPB, 1)
void lstm_persistent(const float* __restrict__ x,      // [B,T,F]
                     const float* __restrict__ Wx,     // [F,4U]
                     const float* __restrict__ bias,   // [4U]
                     const __bf16* __restrict__ whp,   // packed Wh
                     __bf16* __restrict__ hbuf,        // [2][B*U] bf16
                     unsigned* __restrict__ bar,
                     float* __restrict__ out)          // ys | h_T | c_T
{
    const int w    = blockIdx.x;            // workgroup -> units [16w, 16w+16)
    const int lane = threadIdx.x & 31;
    const int wave = threadIdx.x >> 5;      // 0..3 -> rows [32*wave, 32*wave+32)
    const int nloc = lane & 15;
    const int hi   = lane >> 4;             // lane half (A/B layout + C row base)
    const int unit = w * 16 + nloc;

    // --- B fragments register-resident for all 2048 steps: [gate][ktile] ---
    v16bf Bf[4][KT];
    {
        const __bf16* base = whp + (size_t)w * 4 * KT * 512;
        #pragma unroll
        for (int gt = 0; gt < 4; ++gt)
            #pragma unroll
            for (int kt = 0; kt < KT; ++kt) {
                const __bf16* frag = base + ((gt * KT + kt) * 512 + lane * 16);
                union { v16bf v; v8bf h[2]; } u;
                u.h[0] = *(const v8bf*)(frag);
                u.h[1] = *(const v8bf*)(frag + 8);
                Bf[gt][kt] = u.v;
            }
    }

    // --- per-lane Wx column + bias (constant across steps) ---
    float wxr[4][3], bg[4];
    #pragma unroll
    for (int gt = 0; gt < 4; ++gt) {
        int col = gt * U_ + unit;
        wxr[gt][0] = Wx[0 * G_ + col];
        wxr[gt][1] = Wx[1 * G_ + col];
        wxr[gt][2] = Wx[2 * G_ + col];
        bg[gt] = bias[col];
    }

    float cst[2][8];
    #pragma unroll
    for (int mt = 0; mt < 2; ++mt)
        #pragma unroll
        for (int j = 0; j < 8; ++j) cst[mt][j] = 0.0f;

    const int mbase = wave * 32;

    for (int t = 0; t < T_; ++t) {
        const __bf16* hsrc = hbuf + (size_t)(t & 1) * (B_ * U_);
        __bf16*       hdst = hbuf + (size_t)((t + 1) & 1) * (B_ * U_);

        #pragma unroll
        for (int mt = 0; mt < 2; ++mt) {
            const int rtile = mbase + mt * 16;

            // acc = b + x_t @ Wx  (K=3, scalar per C element)
            v8f acc[4];
            #pragma unroll
            for (int j = 0; j < 8; ++j) {
                int r = rtile + j + hi * 8;
                const float* xr = x + ((size_t)r * T_ + t) * F_;
                float x0 = xr[0], x1 = xr[1], x2 = xr[2];
                #pragma unroll
                for (int gt = 0; gt < 4; ++gt)
                    acc[gt][j] = bg[gt] + x0 * wxr[gt][0] + x1 * wxr[gt][1]
                                        + x2 * wxr[gt][2];
            }

            // acc += h @ Wh  (bf16 WMMA, K=256 in 8 tiles of 32)
            const __bf16* arow = hsrc + (size_t)(rtile + nloc) * U_;
            #pragma unroll
            for (int kt = 0; kt < KT; ++kt) {
                union { v16bf v; v8bf h[2]; } a;
                int k0 = kt * 32 + hi * 16;    // A layout: K j..j+15 contiguous
                a.h[0] = *(const v8bf*)(arow + k0);
                a.h[1] = *(const v8bf*)(arow + k0 + 8);
                #pragma unroll
                for (int gt = 0; gt < 4; ++gt)
                    acc[gt] = __builtin_amdgcn_wmma_f32_16x16x32_bf16(
                        false, a.v, false, Bf[gt][kt], (short)0, acc[gt],
                        false, false);
            }

            // gates + state update; C layout: row = rtile + j + 8*hi, col = nloc
            #pragma unroll
            for (int j = 0; j < 8; ++j) {
                int r = rtile + j + hi * 8;
                float ig = fast_sigmoid(acc[0][j]);
                float fg = fast_sigmoid(acc[1][j]);
                float gg = fast_tanh(acc[2][j]);
                float og = fast_sigmoid(acc[3][j]);
                float c  = fg * cst[mt][j] + ig * gg;
                cst[mt][j] = c;
                float h = og * fast_tanh(c);

                __builtin_nontemporal_store(
                    h, &out[((size_t)r * T_ + t) * U_ + unit]);   // ys[b,t,u]
                hdst[(size_t)r * U_ + unit] = (__bf16)h;
                if (t == T_ - 1) {
                    size_t o = (size_t)B_ * T_ * U_;
                    out[o + (size_t)r * U_ + unit] = h;                   // h_T
                    out[o + (size_t)B_ * U_ + (size_t)r * U_ + unit] = c; // c_T
                }
            }
        }

        // ---- cross-WG step barrier (all 16 h slices visible) ----
        __threadfence();
        __syncthreads();
        if (threadIdx.x == 0) {
            __hip_atomic_fetch_add(bar, 1u, __ATOMIC_RELEASE,
                                   __HIP_MEMORY_SCOPE_AGENT);
            unsigned tgt = (unsigned)(NWG * (t + 1));
            while (__hip_atomic_load(bar, __ATOMIC_ACQUIRE,
                                     __HIP_MEMORY_SCOPE_AGENT) < tgt)
                __builtin_amdgcn_s_sleep(1);
        }
        __syncthreads();
    }
}

extern "C" void kernel_launch(void* const* d_in, const int* in_sizes, int n_in,
                              void* d_out, int out_size, void* d_ws, size_t ws_size,
                              hipStream_t stream) {
    const float* x  = (const float*)d_in[0];   // [B,T,F]
    const float* Wx = (const float*)d_in[1];   // [F,4U]
    const float* Wh = (const float*)d_in[2];   // [U,4U]
    const float* b  = (const float*)d_in[3];   // [4U]
    float* out = (float*)d_out;

    char* ws = (char*)d_ws;
    __bf16*   whp  = (__bf16*)ws;                                 // 512 KB
    __bf16*   hbuf = (__bf16*)(ws + 512 * 1024);                  // 128 KB
    unsigned* bar  = (unsigned*)(ws + 512 * 1024 + 128 * 1024);   // 4 B

    pack_Wh<<<1024, 256, 0, stream>>>(Wh, whp);
    init_ws<<<128, 256, 0, stream>>>(hbuf, bar);
    lstm_persistent<<<NWG, TPB, 0, stream>>>(x, Wx, b, whp, hbuf, bar, out);
}
